// L6_77206332113746
// MI455X (gfx1250) — matrix operand order repaired and checked by reading, not compiled, not run on hardware
//
#include <hip/hip_runtime.h>
#include <hip/hip_bf16.h>
#include <math.h>

// ---------------------------------------------------------------------------
// Types for CDNA5 WMMA (wave32): 16x16x32 bf16 -> f32 accumulate.
// ---------------------------------------------------------------------------
typedef __attribute__((ext_vector_type(16))) __bf16 v16bf;
typedef __attribute__((ext_vector_type(8)))  float  v8f;

union Frag32B { v16bf v; uint4 q[2]; };

__device__ __forceinline__ float bf2f(unsigned short u) {
    union { unsigned int i; float f; } v; v.i = ((unsigned int)u) << 16; return v.f;
}
__device__ __forceinline__ unsigned short f2bf(float f) {
    union { float f; unsigned int i; } v; v.f = f;
    unsigned int u = v.i;
    unsigned int r = (u + 0x7fffu + ((u >> 16) & 1u)) >> 16;  // round-to-nearest-even
    return (unsigned short)r;
}
__device__ __forceinline__ float elu_f(float x) { return x > 0.0f ? x : expm1f(x); }

#define CEIL_DIV(a, b) (((a) + (b) - 1) / (b))

// ---------------------------------------------------------------------------
// Weight prep: f32 [fin,fout] -> bf16 [fout,fin] (transposed so the WMMA
// B-fragment loads are contiguous in K).
// ---------------------------------------------------------------------------
__global__ void transpose_w_bf16(const float* __restrict__ W, unsigned short* __restrict__ Wt,
                                 int fin, int fout) {
    int gid = blockIdx.x * blockDim.x + threadIdx.x;
    if (gid >= fin * fout) return;
    int o = gid / fin, k = gid % fin;
    Wt[gid] = f2bf(W[(size_t)k * fout + o]);
}

// gW: f32 [6, fin, fout] -> bf16 [6*fout, fin]; row (c*fout+o) holds W[c,:,o].
__global__ void transpose_gw_bf16(const float* __restrict__ W, unsigned short* __restrict__ Wt,
                                  int fin, int fout) {
    int gid = blockIdx.x * blockDim.x + threadIdx.x;
    if (gid >= 6 * fin * fout) return;
    int row = gid / fin, k = gid % fin;      // row = c*fout + o
    int c = row / fout, o = row % fout;
    Wt[gid] = f2bf(W[((size_t)c * fin + k) * fout + o]);
}

// concat(one_hot[N,20], features[N,44]) -> bf16 x[N,64]
__global__ void concat_bf16(const float* __restrict__ oh, const float* __restrict__ ft,
                            unsigned short* __restrict__ x, int N) {
    int gid = blockIdx.x * blockDim.x + threadIdx.x;
    if (gid >= N * 64) return;
    int n = gid >> 6, j = gid & 63;
    float v = (j < 20) ? oh[(size_t)n * 20 + j] : ft[(size_t)n * 44 + (j - 20)];
    x[gid] = f2bf(v);
}

// ---------------------------------------------------------------------------
// WMMA GEMM: C[M,Nc] = act(A[M,K] * Bt[Nc,K]^T + bias), bf16 in, bf16 out.
// One wave per 16x16 output tile; K stepped by 32.
// A frag (ISA 16-bit A 16x32 layout): lane L, M=L&15, half=L>>4,
//   elems 0..7  -> K = half*8 + j,       elems 8..15 -> K = 16 + half*8 + (j-8)
// B frag (same K order as A): lane L, N=L&15,
//   elems 0..7  -> K = half*8 + j,       elems 8..15 -> K = 16 + half*8 + (j-8)
// ---------------------------------------------------------------------------
__global__ void wmma_gemm_bias_act(const unsigned short* __restrict__ A,
                                   const unsigned short* __restrict__ Bt,
                                   const float* __restrict__ bias,
                                   unsigned short* __restrict__ C,
                                   int M, int K, int Nc, int act) {
    int lane = threadIdx.x & 31;
    int wave = threadIdx.x >> 5;
    int tilesN = Nc >> 4;
    int tile = blockIdx.x * (blockDim.x >> 5) + wave;
    int total = (M >> 4) * tilesN;
    if (tile >= total) return;
    int nt = tile % tilesN;
    int mt = tile / tilesN;
    int l15 = lane & 15;
    int half = lane >> 4;

    const unsigned short* arow = A  + (size_t)(mt * 16 + l15) * K + half * 8;
    const unsigned short* brow = Bt + (size_t)(nt * 16 + l15) * K + half * 8;

    v8f acc = {};
    for (int k = 0; k < K; k += 32) {
        Frag32B a, b;
        a.q[0] = *(const uint4*)(arow + k);        // K: half*8 + [0..7]
        a.q[1] = *(const uint4*)(arow + k + 16);   // K: 16 + half*8 + [0..7]
        b.q[0] = *(const uint4*)(brow + k);        // K: half*8 + [0..7]
        b.q[1] = *(const uint4*)(brow + k + 16);   // K: 16 + half*8 + [0..7]
        acc = __builtin_amdgcn_wmma_f32_16x16x32_bf16(
            false, a.v, false, b.v, (short)0, acc, false, false);
    }

    int col = nt * 16 + l15;
    float bv = bias ? bias[col] : 0.0f;
#pragma unroll
    for (int r = 0; r < 8; ++r) {
        float v = acc[r] + bv;                       // D row M = r + 8*half
        if (act) v = elu_f(v);
        C[(size_t)(mt * 16 + r + half * 8) * Nc + col] = f2bf(v);
    }
}

// Scalar GEMM for tiny layers (K < 32): C[M,Nc] = A[M,K] * Bt[Nc,K]^T, bf16.
__global__ void small_gemm_bf16(const unsigned short* __restrict__ A,
                                const unsigned short* __restrict__ Bt,
                                unsigned short* __restrict__ C, int M, int K, int Nc) {
    int gid = blockIdx.x * blockDim.x + threadIdx.x;
    if (gid >= M * Nc) return;
    int n = gid / Nc, o = gid % Nc;
    const unsigned short* a = A  + (size_t)n * K;
    const unsigned short* w = Bt + (size_t)o * K;
    float s = 0.0f;
    for (int k = 0; k < K; ++k) s += bf2f(a[k]) * bf2f(w[k]);
    C[gid] = f2bf(s);
}

// accum[n,f] = bias[f]
__global__ void bias_init(float* __restrict__ acc, const float* __restrict__ b, int N, int F) {
    int gid = blockIdx.x * blockDim.x + threadIdx.x;
    if (gid >= N * F) return;
    acc[gid] = b[gid % F];
}

// Channel-fused edge scatter: one thread per (edge, 4-wide output quad).
//   accum[dst, f] += sum_c ew[e,c] * H[src, c*Fout + f]
// H rows are L2-resident bf16; atomics are f32 GLOBAL_ATOMIC_ADD_F32.
__global__ void edge_scatter(const unsigned short* __restrict__ H,
                             const int* __restrict__ src, const int* __restrict__ dst,
                             const float* __restrict__ ew, float* __restrict__ accum,
                             long long E, int Fout, int qPerEdge) {
    long long gid = (long long)blockIdx.x * blockDim.x + threadIdx.x;
    long long tot = E * (long long)qPerEdge;
    if (gid >= tot) return;
    long long e = gid / qPerEdge;
    int q = (int)(gid % qPerEdge);
    int s = src[e], d = dst[e];
    const float* w = ew + e * 6;
    const unsigned short* hrow = H + (size_t)s * (6 * Fout) + 4 * q;
    float a0 = 0.f, a1 = 0.f, a2 = 0.f, a3 = 0.f;
#pragma unroll
    for (int c = 0; c < 6; ++c) {
        float wc = w[c];
        uint2 pk = *(const uint2*)(hrow + (size_t)c * Fout);   // 4 bf16
        a0 += wc * bf2f((unsigned short)(pk.x & 0xffff));
        a1 += wc * bf2f((unsigned short)(pk.x >> 16));
        a2 += wc * bf2f((unsigned short)(pk.y & 0xffff));
        a3 += wc * bf2f((unsigned short)(pk.y >> 16));
    }
    float* arow = accum + (size_t)d * Fout + 4 * q;
    atomicAdd(arow + 0, a0);
    atomicAdd(arow + 1, a1);
    atomicAdd(arow + 2, a2);
    atomicAdd(arow + 3, a3);
}

// ELU over accum; write bf16 activation (and optional f32 copy for the head).
__global__ void elu_store(const float* __restrict__ in, unsigned short* __restrict__ obf,
                          float* __restrict__ of32, int n) {
    int gid = blockIdx.x * blockDim.x + threadIdx.x;
    if (gid >= n) return;
    float v = elu_f(in[gid]);
    obf[gid] = f2bf(v);
    if (of32) of32[gid] = v;
}

// Per-node head MLP 4->8->16->8->4->1 (+sigmoid), all in registers.
__global__ void head_kernel(const float* __restrict__ x,
                            const float* __restrict__ W1, const float* __restrict__ b1,
                            const float* __restrict__ W2, const float* __restrict__ b2,
                            const float* __restrict__ W3, const float* __restrict__ b3,
                            const float* __restrict__ W4, const float* __restrict__ b4,
                            const float* __restrict__ W5, const float* __restrict__ b5,
                            float* __restrict__ out, int N) {
    int n = blockIdx.x * blockDim.x + threadIdx.x;
    if (n >= N) return;
    float x4[4];
#pragma unroll
    for (int i = 0; i < 4; ++i) x4[i] = x[(size_t)n * 4 + i];
    float t8[8];
    for (int o = 0; o < 8; ++o) { float s = b1[o]; for (int k = 0; k < 4;  ++k) s += x4[k]  * W1[k * 8  + o]; t8[o]  = elu_f(s); }
    float t16[16];
    for (int o = 0; o < 16; ++o){ float s = b2[o]; for (int k = 0; k < 8;  ++k) s += t8[k]  * W2[k * 16 + o]; t16[o] = elu_f(s); }
    float u8[8];
    for (int o = 0; o < 8; ++o) { float s = b3[o]; for (int k = 0; k < 16; ++k) s += t16[k] * W3[k * 8  + o]; u8[o]  = elu_f(s); }
    float t4[4];
    for (int o = 0; o < 4; ++o) { float s = b4[o]; for (int k = 0; k < 8;  ++k) s += u8[k]  * W4[k * 4  + o]; t4[o]  = elu_f(s); }
    float z = b5[0];
    for (int k = 0; k < 4; ++k) z += t4[k] * W5[k];
    out[n] = 1.0f / (1.0f + expf(-z));
}

// ---------------------------------------------------------------------------
extern "C" void kernel_launch(void* const* d_in, const int* in_sizes, int n_in,
                              void* d_out, int out_size, void* d_ws, size_t ws_size,
                              hipStream_t stream) {
    const int N = in_sizes[2];   // gemme_features has N elements
    const int E = in_sizes[3];

    const float* one_hot  = (const float*)d_in[0];
    const float* features = (const float*)d_in[1];
    const int*   esrc     = (const int*)d_in[3];
    const int*   edst     = (const int*)d_in[4];
    const float* ew       = (const float*)d_in[5];

    const float* eW[4] = {(const float*)d_in[6],  (const float*)d_in[8],
                          (const float*)d_in[10], (const float*)d_in[12]};
    const float* eb[4] = {(const float*)d_in[7],  (const float*)d_in[9],
                          (const float*)d_in[11], (const float*)d_in[13]};
    const float* gW[6]; const float* gb[6];
    for (int i = 0; i < 6; ++i) { gW[i] = (const float*)d_in[14 + 2 * i]; gb[i] = (const float*)d_in[15 + 2 * i]; }
    const float* fw[5]; const float* fb[5];
    for (int i = 0; i < 5; ++i) { fw[i] = (const float*)d_in[26 + 2 * i]; fb[i] = (const float*)d_in[27 + 2 * i]; }

    const int eFin[4]  = {64, 32, 64, 128};
    const int eFout[4] = {32, 64, 128, 256};
    const int gFin[6]  = {256, 128, 64, 32, 16, 8};
    const int gFout[6] = {128, 64, 32, 16, 8, 4};

    // Carve workspace (256B aligned chunks).
    char* p = (char*)d_ws;
    auto carve = [&](size_t bytes) -> void* {
        void* r = (void*)p;
        p += (bytes + 255) & ~(size_t)255;
        return r;
    };
    unsigned short* eWt[4];
    for (int i = 0; i < 4; ++i) eWt[i] = (unsigned short*)carve((size_t)eFin[i] * eFout[i] * 2);
    unsigned short* gWt[6];
    for (int i = 0; i < 6; ++i) gWt[i] = (unsigned short*)carve((size_t)6 * gFin[i] * gFout[i] * 2);
    unsigned short* xA    = (unsigned short*)carve((size_t)N * 256 * 2);
    unsigned short* xB    = (unsigned short*)carve((size_t)N * 256 * 2);
    unsigned short* H     = (unsigned short*)carve((size_t)N * 768 * 2);   // bf16, L2-resident
    float*          accum = (float*)carve((size_t)N * 128 * 4);
    float*          xf4   = (float*)carve((size_t)N * 4 * 4);

    const int TB = 256;  // 8 waves (wave32)

    // Weight conversion + transpose (cheap; every launch for determinism).
    for (int i = 0; i < 4; ++i) {
        int tot = eFin[i] * eFout[i];
        transpose_w_bf16<<<CEIL_DIV(tot, TB), TB, 0, stream>>>(eW[i], eWt[i], eFin[i], eFout[i]);
    }
    for (int i = 0; i < 6; ++i) {
        int tot = 6 * gFin[i] * gFout[i];
        transpose_gw_bf16<<<CEIL_DIV(tot, TB), TB, 0, stream>>>(gW[i], gWt[i], gFin[i], gFout[i]);
    }

    // Encoder input.
    concat_bf16<<<CEIL_DIV(N * 64, TB), TB, 0, stream>>>(one_hot, features, xA, N);

    // Encoder: 4 WMMA GEMM + bias + ELU layers (bf16 activations).
    unsigned short* cur = xA; unsigned short* nxt = xB;
    for (int i = 0; i < 4; ++i) {
        int tiles = (N / 16) * (eFout[i] / 16);
        wmma_gemm_bias_act<<<CEIL_DIV(tiles, 8), TB, 0, stream>>>(
            cur, eWt[i], eb[i], nxt, N, eFin[i], eFout[i], /*act=*/1);
        unsigned short* t = cur; cur = nxt; nxt = t;
    }

    // 6 graph-conv layers.
    for (int i = 0; i < 6; ++i) {
        int fin = gFin[i], fo = gFout[i], F6 = 6 * fo;
        // H = x @ [W_0|...|W_5]  -> [N, 6*fo] bf16
        if (fin >= 32) {
            int tiles = (N / 16) * (F6 / 16);
            wmma_gemm_bias_act<<<CEIL_DIV(tiles, 8), TB, 0, stream>>>(
                cur, gWt[i], nullptr, H, N, fin, F6, /*act=*/0);
        } else {
            small_gemm_bf16<<<CEIL_DIV(N * F6, TB), TB, 0, stream>>>(cur, gWt[i], H, N, fin, F6);
        }
        // accum = bias broadcast, then channel-fused atomic scatter over edges.
        bias_init<<<CEIL_DIV(N * fo, TB), TB, 0, stream>>>(accum, gb[i], N, fo);
        int qpe = fo / 4;
        long long tot = (long long)E * qpe;
        edge_scatter<<<(unsigned)CEIL_DIV(tot, (long long)TB), TB, 0, stream>>>(
            H, esrc, edst, ew, accum, (long long)E, fo, qpe);
        // ELU -> next bf16 activation (+ f32 copy on the last layer for the head).
        elu_store<<<CEIL_DIV(N * fo, TB), TB, 0, stream>>>(
            accum, nxt, (i == 5) ? xf4 : nullptr, N * fo);
        unsigned short* t = cur; cur = nxt; nxt = t;
    }

    // Head MLP + sigmoid.
    head_kernel<<<CEIL_DIV(N, TB), TB, 0, stream>>>(
        xf4, fw[0], fb[0], fw[1], fb[1], fw[2], fb[2], fw[3], fb[3], fw[4], fb[4],
        (float*)d_out, N);
}
